// CSNN_66907000537820
// MI455X (gfx1250) — matrix operand order repaired and checked by reading, not compiled, not run on hardware
//
#include <hip/hip_runtime.h>
#include <hip/hip_bf16.h>

typedef _Float16 h16;
typedef __attribute__((ext_vector_type(16))) _Float16 v16h;
typedef __attribute__((ext_vector_type(8)))  _Float16 v8h;
typedef __attribute__((ext_vector_type(8)))  float    v8f;
typedef __attribute__((ext_vector_type(4)))  float    v4f;

#define TSTEPS 8
#define NT 4   // N-subtiles per wave: each wave computes 16(M) x 64(N)

// ---------------------------------------------------------------------------
// zero-fill helper (for the OOB zero-pad region)
// ---------------------------------------------------------------------------
__global__ void snn_zerofill(h16* __restrict__ p, int n) {
  int i = blockIdx.x * blockDim.x + threadIdx.x;
  if (i < n) p[i] = (h16)0;
}

// ---------------------------------------------------------------------------
// weight permute + f32->f16: w[co,ci,ky,kx] -> wt[co,ky,kx,ci], co < Mpad
// (rows co >= CO are zero-filled so the GEMM A-loads need no guard)
// ---------------------------------------------------------------------------
__global__ void snn_wperm(const float* __restrict__ w, h16* __restrict__ wt,
                          int CO, int Mpad, int CI, int KH, int KW) {
  int idx = blockIdx.x * blockDim.x + threadIdx.x;
  int total = Mpad * KH * KW * CI;
  if (idx >= total) return;
  int ci = idx % CI; int t = idx / CI;
  int kx = t % KW;   t /= KW;
  int ky = t % KH;   int co = t / KH;
  float v = (co < CO) ? w[((co * CI + ci) * KH + ky) * KW + kx] : 0.f;
  wt[idx] = (h16)v;
}

// ---------------------------------------------------------------------------
// conv1: x[32,1,28,28] * w[128,1,5,5] + b, s1 p1 -> y NHWC [32,26,26,128]
// (runs once; input is identical across timesteps)
// ---------------------------------------------------------------------------
__global__ void snn_conv1(const float* __restrict__ x,
                          const float* __restrict__ w,
                          const float* __restrict__ b,
                          float* __restrict__ y) {
  const int OH = 26, OW = 26, IH = 28, IW = 28, CO = 128;
  int idx = blockIdx.x * blockDim.x + threadIdx.x;
  int total = 32 * OH * OW * CO;
  if (idx >= total) return;
  int co = idx & (CO - 1); int t = idx / CO;
  int ox = t % OW;          t /= OW;
  int oy = t % OH;          int bb = t / OH;
  float s = b[co];
#pragma unroll
  for (int ky = 0; ky < 5; ky++) {
    int iy = oy - 1 + ky;
    if ((unsigned)iy >= (unsigned)IH) continue;
#pragma unroll
    for (int kx = 0; kx < 5; kx++) {
      int ix = ox - 1 + kx;
      if ((unsigned)ix >= (unsigned)IW) continue;
      s += x[(bb * IH + iy) * IW + ix] * w[(co * 5 + ky) * 5 + kx];
    }
  }
  y[idx] = s;
}

// ---------------------------------------------------------------------------
// IF node scans (hard reset, v_reset = 0); pre-activations and spikes are
// both NHWC, t-major with stride NB, so these are pure elementwise scans.
// ---------------------------------------------------------------------------
__global__ void snn_if_const(const float* __restrict__ a, h16* __restrict__ s,
                             float vth, int NB) {
  int j = blockIdx.x * blockDim.x + threadIdx.x;
  if (j >= NB) return;
  float inj = a[j];
  float v = 0.f;
#pragma unroll
  for (int t = 0; t < TSTEPS; t++) {
    v += inj;
    float sp = (v >= vth) ? 1.f : 0.f;
    v *= (1.f - sp);
    s[(size_t)t * NB + j] = (h16)sp;
  }
}

__global__ void snn_if_scan(const float* __restrict__ z, h16* __restrict__ s,
                            float vth, int NB) {
  int j = blockIdx.x * blockDim.x + threadIdx.x;
  if (j >= NB) return;
  float v = 0.f;
#pragma unroll
  for (int t = 0; t < TSTEPS; t++) {
    v += z[(size_t)t * NB + j];
    float sp = (v >= vth) ? 1.f : 0.f;
    v *= (1.f - sp);
    s[(size_t)t * NB + j] = (h16)sp;
  }
}

// ---------------------------------------------------------------------------
// NHWC implicit-GEMM conv / FC via v_wmma_f32_16x16x32_f16.
//   W  : [Mpad, KH, KW, Cin] f16   (row-major, K = KH*KW*Cin)
//   S  : [TB, IH, IW, Cin]   f16   (spikes)
//   Z  : [TB, OH, OW, M]     f32
// One wave per 16(M) x 64(N) tile.  (ky,kx) slabs outer, channels inner.
// Per slab each subtile resolves to a 32-bit element OFFSET from S (OOB ->
// zero-pad offset with step 0), so every B access is a global_load_b128 with
// saddr=S + signed voffset and the inner loop is branch-free:
// 2 A loads + 4x(2 B loads + 32-bit bump) + 4 WMMAs per K-step of 32.
// ---------------------------------------------------------------------------
__global__ void __launch_bounds__(32)
snn_wmma_gemm(const h16* __restrict__ W, const h16* __restrict__ S,
              int zoff, const float* __restrict__ bias, float* __restrict__ Z,
              int M, int K, int Cin, int IH, int IW,
              int OH, int OW, int KH, int KW, int strd, int pad, int Ntot) {
  const int ntiles = Ntot / (16 * NT);
  const int mtile = blockIdx.x / ntiles;
  const int ntile = blockIdx.x % ntiles;
  const int mbase = mtile * 16;
  const int nbase0 = ntile * (16 * NT);

  const int lane = threadIdx.x;
  const int lo = lane & 15;
  const int hi = lane >> 4;

  // per-lane output coordinates for each N-subtile (computed once)
  int n_[NT], p0_[NT], q0_[NT], sb_[NT];
#pragma unroll
  for (int j = 0; j < NT; j++) {
    int n = nbase0 + j * 16 + lo;
    int ox = n % OW; int t = n / OW;
    int oy = t % OH; int tb = t / OH;
    n_[j] = n;
    sb_[j] = tb * IH * IW * Cin + hi * 16;  // NHWC base offset (elements)
    p0_[j] = oy * strd - pad;
    q0_[j] = ox * strd - pad;
  }

  v8f acc[NT];
#pragma unroll
  for (int j = 0; j < NT; j++)
#pragma unroll
    for (int r = 0; r < 8; r++) acc[j][r] = 0.f;

  int woff = (mbase + lo) * K + hi * 16;

  for (int ky = 0; ky < KH; ky++) {
    for (int kx = 0; kx < KW; kx++) {
      // ---- per-slab subtile offsets: OOB -> zero-pad offset, zero step ----
      int boff[NT], step[NT];
#pragma unroll
      for (int j = 0; j < NT; j++) {
        int iy = p0_[j] + ky;
        int ix = q0_[j] + kx;
        bool valid = (unsigned)iy < (unsigned)IH && (unsigned)ix < (unsigned)IW;
        boff[j] = valid ? (sb_[j] + (iy * IW + ix) * Cin) : zoff;
        step[j] = valid ? 32 : 0;
      }
      // ---- branch-free channel loop ----
      for (int cc = 0; cc < Cin; cc += 32) {
        // A fragment (contiguous k = 16*hi + s, matching B): two 16B loads
        v8h a_lo = *(const v8h*)(W + woff);
        v8h a_hi = *(const v8h*)(W + woff + 8);
        woff += 32;
        v16h a;
#pragma unroll
        for (int i = 0; i < 8; i++) { a[i] = a_lo[i]; a[8 + i] = a_hi[i]; }

        v16h bfrag[NT];
#pragma unroll
        for (int j = 0; j < NT; j++) {
          v8h b0 = *(const v8h*)(S + boff[j]);
          v8h b1 = *(const v8h*)(S + boff[j] + 8);
          boff[j] += step[j];
#pragma unroll
          for (int i = 0; i < 8; i++) {
            bfrag[j][i] = b0[i];
            bfrag[j][8 + i] = b1[i];
          }
        }

#pragma unroll
        for (int j = 0; j < NT; j++)
          acc[j] = __builtin_amdgcn_wmma_f32_16x16x32_f16(
              false, a, false, bfrag[j], (short)0, acc[j], false, false);
      }
    }
  }

  // ---- epilogue: D rows = 8 consecutive Cout per lane -> NHWC stores ----
  float bv[8];
#pragma unroll
  for (int r = 0; r < 8; r++) {
    int mm = mbase + hi * 8 + r;
    bv[r] = (bias && mm < M) ? bias[mm] : 0.f;
  }
  if ((M & 7) == 0) {       // all real layers: vectorized 2x b128 stores
#pragma unroll
    for (int j = 0; j < NT; j++) {
      int zb = n_[j] * M + mbase + hi * 8;
      v4f s0, s1;
#pragma unroll
      for (int r = 0; r < 4; r++) {
        s0[r] = acc[j][r] + bv[r];
        s1[r] = acc[j][4 + r] + bv[4 + r];
      }
      *(v4f*)(Z + zb) = s0;
      *(v4f*)(Z + zb + 4) = s1;
    }
  } else {                  // fc3 (M=10): guarded scalar stores
#pragma unroll
    for (int j = 0; j < NT; j++) {
#pragma unroll
      for (int r = 0; r < 8; r++) {
        int mm = mbase + hi * 8 + r;
        if (mm < M) Z[n_[j] * M + mm] = acc[j][r] + bv[r];
      }
    }
  }
}

// ---------------------------------------------------------------------------
// firing-rate mean over T: out[b*10+o] = (1/8) sum_t s[t,b,o]
// ---------------------------------------------------------------------------
__global__ void snn_mean(const h16* __restrict__ s, float* __restrict__ out,
                         int NB) {
  int j = blockIdx.x * blockDim.x + threadIdx.x;
  if (j >= NB) return;
  float acc = 0.f;
#pragma unroll
  for (int t = 0; t < TSTEPS; t++) acc += (float)s[(size_t)t * NB + j];
  out[j] = acc * 0.125f;
}

// ---------------------------------------------------------------------------
extern "C" void kernel_launch(void* const* d_in, const int* in_sizes, int n_in,
                              void* d_out, int out_size, void* d_ws,
                              size_t ws_size, hipStream_t stream) {
  (void)in_sizes; (void)n_in; (void)out_size; (void)ws_size;

  const float* x   = (const float*)d_in[0];
  const float* w1  = (const float*)d_in[1];
  const float* b1  = (const float*)d_in[2];
  const float* w2  = (const float*)d_in[3];
  const float* b2  = (const float*)d_in[4];
  const float* w3  = (const float*)d_in[5];
  const float* b3  = (const float*)d_in[6];
  const float* w4  = (const float*)d_in[7];
  const float* b4  = (const float*)d_in[8];
  const float* w5  = (const float*)d_in[9];
  const float* b5  = (const float*)d_in[10];
  const float* fc1 = (const float*)d_in[11];
  const float* fc2 = (const float*)d_in[12];
  const float* fc3 = (const float*)d_in[13];
  float* out = (float*)d_out;

  // ---- workspace carve-up (256B aligned) ----
  size_t off = 0;
  auto carve = [&](size_t bytes) -> void* {
    void* p = (char*)d_ws + off;
    off += (bytes + 255) & ~(size_t)255;
    return p;
  };
  const size_t NW2 = 128 * 16 * 128, NW3 = 192 * 9 * 128, NW4 = 128 * 9 * 192,
               NW5 = 128 * 9 * 128, NF1 = (size_t)2048 * 36 * 128,
               NF2 = (size_t)512 * 2048, NF3 = 16 * 512;  // fc3 padded to 16
  h16* zpad = (h16*)carve(256);  // 128 zero halves for OOB B-loads
  h16* wt2 = (h16*)carve(NW2 * 2);
  h16* wt3 = (h16*)carve(NW3 * 2);
  h16* wt4 = (h16*)carve(NW4 * 2);
  h16* wt5 = (h16*)carve(NW5 * 2);
  h16* wtf1 = (h16*)carve(NF1 * 2);
  h16* wtf2 = (h16*)carve(NF2 * 2);
  h16* wtf3 = (h16*)carve(NF3 * 2);

  const size_t NB1 = 32 * 26 * 26 * 128;   // conv1 out per timestep (NHWC)
  const size_t NB2 = 32 * 13 * 13 * 128;
  const size_t NB3 = 32 * 13 * 13 * 192;
  const size_t NB5 = 32 * 6 * 6 * 128;
  const size_t NB6 = 32 * 2048, NB7 = 32 * 512, NB8 = 32 * 10;

  float* a1 = (float*)carve(NB1 * 4);
  h16* sA = (h16*)carve((size_t)TSTEPS * NB1 * 2);   // largest odd-layer spikes
  h16* sB = (h16*)carve((size_t)TSTEPS * NB2 * 2);   // largest even-layer spikes
  float* z = (float*)carve((size_t)TSTEPS * NB3 * 4);// largest pre-activation

  snn_zerofill<<<1, 128, 0, stream>>>(zpad, 128);

  auto perm = [&](const float* src, h16* dst, int CO, int Mpad, int CI, int KH,
                  int KW) {
    int total = Mpad * KH * KW * CI;
    snn_wperm<<<(total + 255) / 256, 256, 0, stream>>>(src, dst, CO, Mpad, CI,
                                                       KH, KW);
  };
  perm(w2, wt2, 128, 128, 128, 4, 4);
  perm(w3, wt3, 192, 192, 128, 3, 3);
  perm(w4, wt4, 128, 128, 192, 3, 3);
  perm(w5, wt5, 128, 128, 128, 3, 3);
  perm(fc1, wtf1, 2048, 2048, 128, 6, 6);  // [2048,128,6,6] -> [2048,6,6,128]
  perm(fc2, wtf2, 512, 512, 2048, 1, 1);
  perm(fc3, wtf3, 10, 16, 512, 1, 1);      // zero-padded rows 10..15

  // layer 1 (run once, input constant over T) + IF scan
  {
    int total = (int)NB1;
    snn_conv1<<<(total + 255) / 256, 256, 0, stream>>>(x, w1, b1, a1);
    snn_if_const<<<(total + 255) / 256, 256, 0, stream>>>(a1, sA, 13515.0f,
                                                          total);
  }

  auto gemm = [&](const h16* W, const h16* S, const float* bias, float* Z,
                  int M, int Mpad, int Cin, int IH, int IW, int OH, int OW,
                  int KH, int KW, int strd, int pad) {
    int K = KH * KW * Cin;
    int Ntot = TSTEPS * 32 * OH * OW;       // merged T*B batch = 256
    int mtiles = Mpad / 16;
    int ntiles = Ntot / (16 * NT);
    int zoff = (int)(zpad - S);             // zero-pad as element offset from S
    snn_wmma_gemm<<<mtiles * ntiles, 32, 0, stream>>>(
        W, S, zoff, bias, Z, M, K, Cin, IH, IW, OH, OW, KH, KW, strd, pad,
        Ntot);
  };
  auto ifscan = [&](const float* Z, h16* S, float vth, size_t NB) {
    snn_if_scan<<<(int)((NB + 255) / 256), 256, 0, stream>>>(Z, S, vth,
                                                             (int)NB);
  };

  // conv2: 128->128, 4x4, s2, p1 : 26 -> 13
  gemm(wt2, sA, b2, z, 128, 128, 128, 26, 26, 13, 13, 4, 4, 2, 1);
  ifscan(z, sB, 71.0f, NB2);
  // conv3: 128->192, 3x3, s1, p1 : 13 -> 13
  gemm(wt3, sB, b3, z, 192, 192, 128, 13, 13, 13, 13, 3, 3, 1, 1);
  ifscan(z, sA, 93.0f, NB3);
  // conv4: 192->128, 3x3, s1, p1 : 13 -> 13
  gemm(wt4, sA, b4, z, 128, 128, 192, 13, 13, 13, 13, 3, 3, 1, 1);
  ifscan(z, sB, 144.0f, NB2);
  // conv5: 128->128, 3x3, s2, p0 : 13 -> 6
  gemm(wt5, sB, b5, z, 128, 128, 128, 13, 13, 6, 6, 3, 3, 2, 0);
  ifscan(z, sA, 79.0f, NB5);
  // fc1: (6*6*128 NHWC) -> 2048, weights permuted to match
  gemm(wtf1, sA, nullptr, z, 2048, 2048, 128, 6, 6, 1, 1, 6, 6, 1, 0);
  ifscan(z, sB, 2475.0f, NB6);
  // fc2: 2048 -> 512
  gemm(wtf2, sB, nullptr, z, 512, 512, 2048, 1, 1, 1, 1, 1, 1, 1, 0);
  ifscan(z, sA, 1357.0f, NB7);
  // fc3: 512 -> 10 (A rows padded with zeros to 16)
  gemm(wtf3, sA, nullptr, z, 10, 16, 512, 1, 1, 1, 1, 1, 1, 1, 0);
  ifscan(z, sB, 388.0f, NB8);

  snn_mean<<<(int)((NB8 + 255) / 256), 256, 0, stream>>>(sB, out, (int)NB8);
}
